// MultiGCN_66812511257310
// MI455X (gfx1250) — matrix operand order repaired and checked by reading, not compiled, not run on hardware
//
#include <hip/hip_runtime.h>
#include <hip/hip_bf16.h>

// MultiGCN on MI455X (gfx1250, wave32).
//
// Pipeline per call (all on `stream`, all scratch in d_ws):
//   1) deg histogram over edge dst (int atomics)
//   2) exclusive scan -> CSR row offsets (single 1024-thread workgroup)
//   3) dinv = rsqrt(deg+1); cursor = offsets copy
//   4) CSR fill (scatter src ids by dst)
//   per layer:
//   5) GEMM xw = feat @ W via v_wmma_f32_16x16x32_bf16 with bf16 hi/lo
//      split (3 WMMAs / k-step) for ~fp32 accuracy
//   6) aggregation: one block per dst node, register accumulation over
//      incoming edges (no float atomics), fused bias (+relu+residual or
//      +log_softmax for the final layer).

typedef __attribute__((ext_vector_type(16))) __bf16 v16bf;
typedef __attribute__((ext_vector_type(8)))  float  v8f;

// ---------------------------------------------------------------- utilities
__global__ void zero_int_kernel(int* __restrict__ p, int n) {
    int i = blockIdx.x * blockDim.x + threadIdx.x;
    if (i < n) p[i] = 0;
}

__global__ void hist_kernel(const int* __restrict__ dst, int e, int* __restrict__ deg) {
    int i = blockIdx.x * blockDim.x + threadIdx.x;
    if (i < e) atomicAdd(&deg[dst[i]], 1);
}

// Exclusive scan of n ints with one 1024-thread workgroup (n ~ 50k: 49 chunks).
__global__ void scan_excl_kernel(const int* __restrict__ in, int* __restrict__ out, int n) {
    __shared__ int s[1024];
    __shared__ int carry_s;
    const int tid = threadIdx.x;
    if (tid == 0) carry_s = 0;
    __syncthreads();
    for (int base = 0; base < n; base += 1024) {
        int i = base + tid;
        int v = (i < n) ? in[i] : 0;
        s[tid] = v;
        __syncthreads();
        for (int off = 1; off < 1024; off <<= 1) {
            int t = (tid >= off) ? s[tid - off] : 0;
            __syncthreads();
            s[tid] += t;
            __syncthreads();
        }
        int incl  = s[tid];
        int carry = carry_s;
        if (i < n) out[i] = carry + incl - v;   // exclusive
        __syncthreads();
        if (tid == 1023) carry_s = carry + s[1023];
        __syncthreads();
    }
    if (tid == 0) out[n] = carry_s;
}

__global__ void dinv_cursor_kernel(const int* __restrict__ deg, const int* __restrict__ offs,
                                   float* __restrict__ dinv, int* __restrict__ cursor, int n) {
    int i = blockIdx.x * blockDim.x + threadIdx.x;
    if (i < n) {
        dinv[i]   = rsqrtf((float)(deg[i] + 1));  // +1 self loop -> always > 0
        cursor[i] = offs[i];
    }
}

__global__ void fill_csr_kernel(const int* __restrict__ src, const int* __restrict__ dst,
                                int e, int* __restrict__ cursor, int* __restrict__ csr_src) {
    int i = blockIdx.x * blockDim.x + threadIdx.x;
    if (i < e) {
        int p = atomicAdd(&cursor[dst[i]], 1);
        csr_src[p] = src[i];
    }
}

// ---------------------------------------------------------------- WMMA GEMM
// out[M x G] = A[M x 128] @ W[128 x G], fp32 in/out, bf16 hi/lo split inside.
// One wave per 16x16 output tile. K fixed at 128.
__global__ __launch_bounds__(32)
void gemm_wmma_kernel(const float* __restrict__ A, const float* __restrict__ W,
                      float* __restrict__ out, int Nrows, int G) {
    const int lane  = threadIdx.x & 31;
    const int row0  = blockIdx.x * 16;
    const int col0  = blockIdx.y * 16;
    const int hiHalf = (lane >= 16);
    const int halfA  = hiHalf ? 8 : 0;      // A: K offset per lane half
    const int halfB  = hiHalf ? 16 : 0;     // unused; B also takes halfA
    int mrow = row0 + (lane & 15);
    if (mrow >= Nrows) mrow = Nrows - 1;    // safe clamp (stores guarded)
    const int ncol = col0 + (lane & 15);

    v8f acc = {};
    #pragma unroll
    for (int kb = 0; kb < 128; kb += 32) {
        v16bf a_hi, a_lo, b_hi, b_lo;
        // A fragment: 16x32 bf16, row = mrow, ISA layout:
        // lanes 0-15:  elems 0-7 -> K=kb+0..7,  elems 8-15 -> K=kb+16..23
        // lanes 16-31: elems 0-7 -> K=kb+8..15, elems 8-15 -> K=kb+24..31
        #pragma unroll
        for (int i = 0; i < 16; ++i) {
            int k = kb + ((i >> 3) << 4) + halfA + (i & 7);
            float f = A[mrow * 128 + k];
            __bf16 h = (__bf16)f;
            a_hi[i] = h;
            a_lo[i] = (__bf16)(f - (float)h);
        }
        // B fragment: 32x16 bf16, N = lane%16, K = (i/8)*16 + (lane/16)*8 + i%8
        #pragma unroll
        for (int i = 0; i < 16; ++i) {
            int k = kb + ((i >> 3) << 4) + halfA + (i & 7);
            float f = (ncol < G) ? W[k * G + ncol] : 0.0f;
            __bf16 h = (__bf16)f;
            b_hi[i] = h;
            b_lo[i] = (__bf16)(f - (float)h);
        }
        // fp32-accuracy split product: Ahi*Bhi + Alo*Bhi + Ahi*Blo
        acc = __builtin_amdgcn_wmma_f32_16x16x32_bf16(false, a_hi, false, b_hi,
                                                      (short)0, acc, false, false);
        acc = __builtin_amdgcn_wmma_f32_16x16x32_bf16(false, a_lo, false, b_hi,
                                                      (short)0, acc, false, false);
        acc = __builtin_amdgcn_wmma_f32_16x16x32_bf16(false, a_hi, false, b_lo,
                                                      (short)0, acc, false, false);
    }
    // C/D layout: VGPR v -> row = row0 + v + 8*lanehalf, col = ncol
    if (ncol < G) {
        #pragma unroll
        for (int v = 0; v < 8; ++v) {
            int r = row0 + v + (hiHalf ? 8 : 0);
            if (r < Nrows) out[r * G + ncol] = acc[v];
        }
    }
}

// ------------------------------------------------------- aggregation kernels
// One block (G=128 threads) per dst node. No atomics: CSR gather + register acc.
__global__ void aggregate_relu_res_kernel(const float* __restrict__ xw,
                                          const float* __restrict__ dinv,
                                          const int* __restrict__ offs,
                                          const int* __restrict__ csr,
                                          const float* __restrict__ bias,
                                          const float* __restrict__ resid,
                                          float* __restrict__ out, int G) {
    const int v = blockIdx.x;
    const int f = threadIdx.x;            // blockDim.x == G == 128
    const float dv = dinv[v];
    float acc = bias[f] + xw[v * G + f] * dv * dv;   // self loop
    const int e0 = offs[v], e1 = offs[v + 1];
    for (int e = e0; e < e1; ++e) {
        int s = csr[e];
        if (e + 1 < e1) __builtin_prefetch(&xw[csr[e + 1] * G + f], 0, 0);
        acc += xw[s * G + f] * (dinv[s] * dv);
    }
    out[v * G + f] = fmaxf(acc, 0.0f) + resid[v * G + f];
}

// Final layer: aggregation + bias + log_softmax fused. blockDim = 64, C <= 64.
__global__ void aggregate_logsoftmax_kernel(const float* __restrict__ xw,
                                            const float* __restrict__ dinv,
                                            const int* __restrict__ offs,
                                            const int* __restrict__ csr,
                                            const float* __restrict__ bias,
                                            float* __restrict__ out, int C) {
    __shared__ float red[64];
    const int v = blockIdx.x;
    const int f = threadIdx.x;            // 0..63
    const float dv = dinv[v];
    float acc = 0.0f;
    if (f < C) {
        acc = bias[f] + xw[v * C + f] * dv * dv;
        const int e0 = offs[v], e1 = offs[v + 1];
        for (int e = e0; e < e1; ++e) {
            int s = csr[e];
            acc += xw[s * C + f] * (dinv[s] * dv);
        }
    }
    // max reduce over C logits
    red[f] = (f < C) ? acc : -3.4e38f;
    __syncthreads();
    for (int off = 32; off > 0; off >>= 1) {
        if (f < off) red[f] = fmaxf(red[f], red[f + off]);
        __syncthreads();
    }
    const float mx = red[0];
    __syncthreads();
    red[f] = (f < C) ? expf(acc - mx) : 0.0f;
    __syncthreads();
    for (int off = 32; off > 0; off >>= 1) {
        if (f < off) red[f] += red[f + off];
        __syncthreads();
    }
    const float lse = logf(red[0]);
    if (f < C) out[v * C + f] = acc - mx - lse;
}

// ------------------------------------------------------------------- launch
extern "C" void kernel_launch(void* const* d_in, const int* in_sizes, int n_in,
                              void* d_out, int out_size, void* d_ws, size_t ws_size,
                              hipStream_t stream) {
    const float* x  = (const float*)d_in[0];
    const int*   ei = (const int*)d_in[1];
    const float* W1 = (const float*)d_in[2];
    const float* b1 = (const float*)d_in[3];
    const float* W2 = (const float*)d_in[4];
    const float* b2 = (const float*)d_in[5];
    const float* W3 = (const float*)d_in[6];
    const float* b3 = (const float*)d_in[7];

    const int N = in_sizes[0] / 128;   // 50000
    const int E = in_sizes[1] / 2;     // 800000
    const int C = in_sizes[7];         // 40
    const int H = 128;

    const int* e_src = ei;
    const int* e_dst = ei + E;

    // workspace partition (256B aligned)
    char* p = (char*)d_ws;
    auto take = [&](size_t bytes) -> void* {
        void* r = (void*)p;
        p += (bytes + 255) & ~(size_t)255;
        return r;
    };
    int*   deg    = (int*)  take((size_t)N * 4);
    int*   offs   = (int*)  take((size_t)(N + 1) * 4);
    int*   cursor = (int*)  take((size_t)N * 4);
    float* dinv   = (float*)take((size_t)N * 4);
    int*   csr    = (int*)  take((size_t)E * 4);
    float* xw     = (float*)take((size_t)N * H * 4);
    float* hA     = (float*)take((size_t)N * H * 4);
    float* hB     = (float*)take((size_t)N * H * 4);
    (void)ws_size; (void)n_in; (void)out_size;

    float* out = (float*)d_out;

    const int TB = 256;
    dim3 gN((N + TB - 1) / TB), gE((E + TB - 1) / TB);

    // graph preprocessing
    zero_int_kernel<<<gN, TB, 0, stream>>>(deg, N);
    hist_kernel<<<gE, TB, 0, stream>>>(e_dst, E, deg);
    scan_excl_kernel<<<1, 1024, 0, stream>>>(deg, offs, N);
    dinv_cursor_kernel<<<gN, TB, 0, stream>>>(deg, offs, dinv, cursor, N);
    fill_csr_kernel<<<gE, TB, 0, stream>>>(e_src, e_dst, E, cursor, csr);

    const int mt = (N + 15) / 16;
    dim3 gemmGridH(mt, H / 16);          // 128-col layers
    dim3 gemmGridC(mt, (C + 15) / 16);   // 40-col layer

    // layer 1
    gemm_wmma_kernel<<<gemmGridH, 32, 0, stream>>>(x, W1, xw, N, H);
    aggregate_relu_res_kernel<<<N, H, 0, stream>>>(xw, dinv, offs, csr, b1, x, hA, H);
    // layer 2
    gemm_wmma_kernel<<<gemmGridH, 32, 0, stream>>>(hA, W2, xw, N, H);
    aggregate_relu_res_kernel<<<N, H, 0, stream>>>(xw, dinv, offs, csr, b2, hA, hB, H);
    // layer 3 + log_softmax
    gemm_wmma_kernel<<<gemmGridC, 32, 0, stream>>>(hB, W3, xw, N, C);
    aggregate_logsoftmax_kernel<<<N, 64, 0, stream>>>(xw, dinv, offs, csr, b3, out, C);
}
